// GCN_layers_88201448391209
// MI455X (gfx1250) — matrix operand order repaired and checked by reading, not compiled, not run on hardware
//
#include <hip/hip_runtime.h>
#include <hip/hip_bf16.h>

// ---------------------------------------------------------------------------
// GATv2 x2 on MI455X (gfx1250, wave32).
// GEMMs ([100000x128]@[128x128] x4) run on v_wmma_f32_16x16x32_f16 (f32 accum),
// with W pre-swizzled into B-fragment order in LDS so every fragment is two
// ds_load_b128 (no scalar u16 LDS loads / half-merges in the hot loop), and A
// fragments loaded as two global_load_b128 per lane.
// Edge phase (1.7M edges x 128 ch, ~3.4GB/layer, L2-resident on 192MB L2) uses
// wave-per-edge gathers + shfl_xor reductions + f32 global atomics.
// ---------------------------------------------------------------------------

#define NNODES 100000
#define NEDGES 1600000
#define EPLUS  (NEDGES + NNODES)   // edges + self loops
#define DIM    128

typedef __attribute__((ext_vector_type(16))) _Float16 v16h;
typedef __attribute__((ext_vector_type(8)))  _Float16 v8h;
typedef __attribute__((ext_vector_type(8)))  float    v8f;

// ---- order-preserving float <-> uint encoding for atomicMax-based segment max
__device__ __forceinline__ unsigned encf(float x) {
    unsigned u = __float_as_uint(x);
    return (u & 0x80000000u) ? ~u : (u | 0x80000000u);
}
__device__ __forceinline__ float decf(unsigned u) {
    unsigned v = (u & 0x80000000u) ? (u & 0x7FFFFFFFu) : ~u;
    return __uint_as_float(v);
}
#define ENC_NEG_INF 0x007FFFFFu   // encf(-inf)

// ---------------------------------------------------------------------------
__global__ void gat_fill_u32(unsigned* __restrict__ p, unsigned v, int n) {
    int i = blockIdx.x * blockDim.x + threadIdx.x;
    if (i < n) p[i] = v;
}

__global__ void gat_cvt_f16(const float* __restrict__ in, _Float16* __restrict__ out, int n) {
    int i = blockIdx.x * blockDim.x + threadIdx.x;
    if (i < n) out[i] = (_Float16)in[i];
}

// out1 accum + bias -> softplus -> f16 activations for layer 2
__global__ void gat_bias_softplus_cvt(const float* __restrict__ acc, const float* __restrict__ bias,
                                      _Float16* __restrict__ out, int n) {
    int i = blockIdx.x * blockDim.x + threadIdx.x;
    if (i >= n) return;
    float v = acc[i] + bias[i & (DIM - 1)];
    float sp = (v > 20.f) ? v : log1pf(__expf(v));
    out[i] = (_Float16)sp;
}

__global__ void gat_add_bias(float* __restrict__ out, const float* __restrict__ bias, int n) {
    int i = blockIdx.x * blockDim.x + threadIdx.x;
    if (i < n) out[i] += bias[i & (DIM - 1)];
}

__global__ void gat_decode_max(unsigned* __restrict__ p, int n) {
    int i = blockIdx.x * blockDim.x + threadIdx.x;
    if (i < n) {
        unsigned u = p[i];
        reinterpret_cast<float*>(p)[i] = decf(u);
    }
}

// ---------------------------------------------------------------------------
// WMMA GEMM: C[M,128] = A_f16[M,128] @ W_f32[128,128] + bias[128]
// 256 threads = 8 waves; wave w owns 16 rows; 8 col tiles x 4 k-steps of
// v_wmma_f32_16x16x32_f16.
//
// Fragment layouts per CDNA5 ISA 7.12.2 (wave32):
//   A 16x32 f16 : lane l -> row l&15, g=l>>4; halves j=0..7 -> k=g*8+j,
//                 halves j=8..15 -> k=16+g*8+(j-8)     (two contiguous 16B runs)
//   B 32x16 f16 : lane l -> col l&15, g=l>>4; half j -> k as for A (j<8: g*8+j, else 16+g*8+(j-8))
//   C 16x16 f32 : lane l -> col l&15; VGPR v -> row (l>>4)*8+v
//
// W is converted f32->f16 and scattered into fragment order in LDS once per
// block: half j of fragment (kk,nt) lane l lives at sW[((kk*8+nt)*32+l)*16+j],
// so the hot loop reads each fragment as two ds_load_b128.
// ---------------------------------------------------------------------------
__global__ __launch_bounds__(256)
void gat_gemm_wmma(const _Float16* __restrict__ A, const float* __restrict__ W,
                   const float* __restrict__ bias, float* __restrict__ C, int M) {
    __shared__ __attribute__((aligned(16))) _Float16 sW[DIM * DIM];  // 32 KB, frag-order

    // one-time scatter: W[k][n] -> fragment (kk=k>>5, nt=n>>4), lane=((k>>3)&1)*16+(n&15),
    // half j = (k&7)|((k>>1)&8)  (same table k order as the A fragment)
    for (int i = threadIdx.x; i < DIM * DIM; i += 256) {
        const int k = i >> 7, n = i & 127;
        const int kk = k >> 5, g = (k >> 3) & 1, j = (k & 7) | ((k >> 1) & 8);
        const int nt = n >> 4, nl = n & 15;
        sW[(((kk * 8 + nt) * 32) + g * 16 + nl) * 16 + j] = (_Float16)W[i];
    }
    __syncthreads();

    const int lane = threadIdx.x & 31;
    const int wave = threadIdx.x >> 5;
    const int rowBase = (blockIdx.x * 8 + wave) * 16;
    const int lmod = lane & 15;
    const int g    = lane >> 4;

    v8f acc[8] = {};   // 8 tiles of 16x16 f32 accumulators

    // clamp row for loads (EXEC must stay all-1 around WMMA); stores guarded
    int mLoad = rowBase + lmod;
    if (mLoad >= M) mLoad = M - 1;
    const _Float16* Arow = A + (size_t)mLoad * DIM;

#pragma unroll
    for (int kk = 0; kk < 4; ++kk) {
        // ---- A fragment: two 16B global loads, no element shuffling needed
        const _Float16* Ap = Arow + kk * 32 + g * 8;
        const v8h aLo = *reinterpret_cast<const v8h*>(Ap);        // k = g*8 + 0..7
        const v8h aHi = *reinterpret_cast<const v8h*>(Ap + 16);   // k = 16 + g*8 + 0..7
        const v16h a = __builtin_shufflevector(aLo, aHi,
                          0, 1, 2, 3, 4, 5, 6, 7, 8, 9, 10, 11, 12, 13, 14, 15);
        // ---- 8 column tiles; each B fragment = two ds_load_b128
        const v8h* fragBase = reinterpret_cast<const v8h*>(sW) + (kk * 8) * 64 + lane * 2;
#pragma unroll
        for (int nt = 0; nt < 8; ++nt) {
            const v8h bLo = fragBase[nt * 64 + 0];
            const v8h bHi = fragBase[nt * 64 + 1];
            const v16h b = __builtin_shufflevector(bLo, bHi,
                              0, 1, 2, 3, 4, 5, 6, 7, 8, 9, 10, 11, 12, 13, 14, 15);
            acc[nt] = __builtin_amdgcn_wmma_f32_16x16x32_f16(
                false, a, false, b, (short)0, acc[nt], false, false);
        }
    }

    // ---- store D tiles (+bias), guard row bound
#pragma unroll
    for (int nt = 0; nt < 8; ++nt) {
        const int n = nt * 16 + lmod;
        const float bn = bias[n];
#pragma unroll
        for (int v = 0; v < 8; ++v) {
            const int m = rowBase + g * 8 + v;
            if (m < M) C[(size_t)m * DIM + n] = acc[nt][v] + bn;
        }
    }
}

// ---------------------------------------------------------------------------
// Edge pass 1: logits[e,h] = sum_c leaky(xl[src]+xr[dst]) * att ; segment max
// One wave32 per edge; lane handles channels {lane, lane+32, lane+64, lane+96}.
// ---------------------------------------------------------------------------
template <int H>
__global__ void gat_edge_logits(const float* __restrict__ xl, const float* __restrict__ xr,
                                const float* __restrict__ att,
                                const int* __restrict__ srcArr, const int* __restrict__ dstArr,
                                float* __restrict__ logits, unsigned* __restrict__ mxEnc) {
    int tid  = blockIdx.x * blockDim.x + threadIdx.x;
    int wid  = tid >> 5;
    int lane = tid & 31;
    if (wid >= EPLUS) return;
    int s, d;
    if (wid < NEDGES) { s = srcArr[wid]; d = dstArr[wid]; }
    else              { s = d = wid - NEDGES; }

    const float* xls = xl + (size_t)s * DIM;
    const float* xrd = xr + (size_t)d * DIM;

    float hp[H] = {};
#pragma unroll
    for (int v = 0; v < 4; ++v) {
        const int ch = v * 32 + lane;
        float gv = xls[ch] + xrd[ch];
        gv = gv > 0.f ? gv : 0.2f * gv;          // leaky relu, slope 0.2
        hp[(H == 4) ? v : 0] += gv * att[ch];    // att flat [H*C] == channel idx
    }
#pragma unroll
    for (int h = 0; h < H; ++h)
#pragma unroll
        for (int off = 16; off; off >>= 1)
            hp[h] += __shfl_xor(hp[h], off, 32);

    if (lane == 0) {
#pragma unroll
        for (int h = 0; h < H; ++h) {
            logits[(size_t)wid * H + h] = hp[h];
            atomicMax(&mxEnc[(size_t)d * H + h], encf(hp[h]));
        }
    }
}

// ---------------------------------------------------------------------------
// Edge pass 2: ex = exp(logit - mx[dst]); logits <- ex; den[dst] += ex
// ---------------------------------------------------------------------------
template <int H>
__global__ void gat_edge_denom(float* __restrict__ logits, const float* __restrict__ mx,
                               float* __restrict__ den, const int* __restrict__ dstArr) {
    int idx = blockIdx.x * blockDim.x + threadIdx.x;
    if (idx >= EPLUS * H) return;
    int e = idx / H;
    int h = idx - e * H;
    int d = (e < NEDGES) ? dstArr[e] : e - NEDGES;
    float ex = __expf(logits[idx] - mx[(size_t)d * H + h]);
    logits[idx] = ex;
    atomicAdd(&den[(size_t)d * H + h], ex);
}

// ---------------------------------------------------------------------------
// Edge pass 3: out[dst] += (ex/den[dst]) * xl[src]   (wave per edge)
// ---------------------------------------------------------------------------
template <int H>
__global__ void gat_edge_aggregate(const float* __restrict__ xl, const float* __restrict__ exv,
                                   const float* __restrict__ den,
                                   const int* __restrict__ srcArr, const int* __restrict__ dstArr,
                                   float* __restrict__ out) {
    int tid  = blockIdx.x * blockDim.x + threadIdx.x;
    int wid  = tid >> 5;
    int lane = tid & 31;
    if (wid >= EPLUS) return;
    int s, d;
    if (wid < NEDGES) { s = srcArr[wid]; d = dstArr[wid]; }
    else              { s = d = wid - NEDGES; }

    float alpha[H];
#pragma unroll
    for (int h = 0; h < H; ++h)
        alpha[h] = exv[(size_t)wid * H + h] / den[(size_t)d * H + h];

    const float* xls = xl + (size_t)s * DIM;
    float*       od  = out + (size_t)d * DIM;
#pragma unroll
    for (int v = 0; v < 4; ++v) {
        const int ch = v * 32 + lane;
        atomicAdd(&od[ch], alpha[(H == 4) ? v : 0] * xls[ch]);
    }
}

// ---------------------------------------------------------------------------
static inline int cdiv(long a, long b) { return (int)((a + b - 1) / b); }

extern "C" void kernel_launch(void* const* d_in, const int* in_sizes, int n_in,
                              void* d_out, int out_size, void* d_ws, size_t ws_size,
                              hipStream_t stream) {
    const float* x    = (const float*)d_in[0];
    const int*   ei   = (const int*)d_in[1];
    const int*   src  = ei;
    const int*   dst  = ei + NEDGES;
    const float* Wl1  = (const float*)d_in[2];
    const float* bl1  = (const float*)d_in[3];
    const float* Wr1  = (const float*)d_in[4];
    const float* br1  = (const float*)d_in[5];
    const float* att1 = (const float*)d_in[6];
    const float* bias1= (const float*)d_in[7];
    const float* Wl2  = (const float*)d_in[8];
    const float* bl2  = (const float*)d_in[9];
    const float* Wr2  = (const float*)d_in[10];
    const float* br2  = (const float*)d_in[11];
    const float* att2 = (const float*)d_in[12];
    const float* bias2= (const float*)d_in[13];
    float* out = (float*)d_out;

    // ---- workspace carve (~210 MB), layer 2 reuses all buffers
    char* ws = (char*)d_ws;
    size_t off = 0;
    auto carve = [&](size_t bytes) { char* p = ws + off; off += (bytes + 255) & ~(size_t)255; return p; };
    _Float16* xh     = (_Float16*)carve((size_t)NNODES * DIM * sizeof(_Float16)); // activations f16
    float*    bufA   = (float*)   carve((size_t)NNODES * DIM * sizeof(float));    // xl
    float*    bufB   = (float*)   carve((size_t)NNODES * DIM * sizeof(float));    // xr
    float*    bufH   = (float*)   carve((size_t)NNODES * DIM * sizeof(float));    // layer1 accum
    float*    logits = (float*)   carve((size_t)EPLUS * 4 * sizeof(float));
    unsigned* mx     = (unsigned*)carve((size_t)NNODES * 4 * sizeof(unsigned));
    float*    den    = (float*)   carve((size_t)NNODES * 4 * sizeof(float));

    const int nElem   = NNODES * DIM;                 // 12.8M
    const int gElem   = cdiv(nElem, 256);
    const int gGemm   = cdiv(NNODES, 128);            // 8 waves x 16 rows / block
    const int gEdgeW  = cdiv((long)EPLUS * 32, 256);  // wave-per-edge kernels

    // ================= Layer 1 (H=4, C=32) =================
    gat_cvt_f16<<<gElem, 256, 0, stream>>>(x, xh, nElem);
    gat_gemm_wmma<<<gGemm, 256, 0, stream>>>(xh, Wl1, bl1, bufA, NNODES);
    gat_gemm_wmma<<<gGemm, 256, 0, stream>>>(xh, Wr1, br1, bufB, NNODES);

    gat_fill_u32<<<cdiv(NNODES * 4, 256), 256, 0, stream>>>(mx, ENC_NEG_INF, NNODES * 4);
    gat_fill_u32<<<cdiv(NNODES * 4, 256), 256, 0, stream>>>((unsigned*)den, 0u, NNODES * 4);
    gat_fill_u32<<<gElem, 256, 0, stream>>>((unsigned*)bufH, 0u, nElem);

    gat_edge_logits<4><<<gEdgeW, 256, 0, stream>>>(bufA, bufB, att1, src, dst, logits, mx);
    gat_decode_max<<<cdiv(NNODES * 4, 256), 256, 0, stream>>>(mx, NNODES * 4);
    gat_edge_denom<4><<<cdiv((long)EPLUS * 4, 256), 256, 0, stream>>>(logits, (float*)mx, den, dst);
    gat_edge_aggregate<4><<<gEdgeW, 256, 0, stream>>>(bufA, logits, den, src, dst, bufH);

    gat_bias_softplus_cvt<<<gElem, 256, 0, stream>>>(bufH, bias1, xh, nElem);

    // ================= Layer 2 (H=1, C=128) =================
    gat_gemm_wmma<<<gGemm, 256, 0, stream>>>(xh, Wl2, bl2, bufA, NNODES);
    gat_gemm_wmma<<<gGemm, 256, 0, stream>>>(xh, Wr2, br2, bufB, NNODES);

    gat_fill_u32<<<cdiv(NNODES, 256), 256, 0, stream>>>(mx, ENC_NEG_INF, NNODES);
    gat_fill_u32<<<cdiv(NNODES, 256), 256, 0, stream>>>((unsigned*)den, 0u, NNODES);
    gat_fill_u32<<<gElem, 256, 0, stream>>>((unsigned*)out, 0u, nElem);

    gat_edge_logits<1><<<gEdgeW, 256, 0, stream>>>(bufA, bufB, att2, src, dst, logits, mx);
    gat_decode_max<<<cdiv(NNODES, 256), 256, 0, stream>>>(mx, NNODES);
    gat_edge_denom<1><<<cdiv((long)EPLUS, 256), 256, 0, stream>>>(logits, (float*)mx, den, dst);
    gat_edge_aggregate<1><<<gEdgeW, 256, 0, stream>>>(bufA, logits, den, src, dst, out);

    gat_add_bias<<<gElem, 256, 0, stream>>>(out, bias2, nElem);

    (void)in_sizes; (void)n_in; (void)out_size; (void)ws_size;
}
